// SelectiveSSM_47390669144437
// MI455X (gfx1250) — matrix operand order repaired and checked by reading, not compiled, not run on hardware
//
#include <hip/hip_runtime.h>

// ---------------- problem constants (from reference) ----------------
#define DM      1024            // d_model
#define DI      2048            // d_inner
#define NST     16              // d_state
#define DTR     64              // dt_rank
#define BB      2               // batch
#define LL      2048            // seq len
#define BL      (BB*LL)         // 4096 tokens
#define XPR     96              // dt_rank + 2*d_state

// ---------------- vector / fragment types ----------------
typedef __attribute__((ext_vector_type(16))) __bf16 v16bf;
typedef __attribute__((ext_vector_type(8)))  __bf16 v8bf;
typedef __attribute__((ext_vector_type(8)))  float  v8f;

union Frag16 { v16bf v; v8bf h[2]; };
union Acc8   { v8f v; float f[8]; };

__device__ __forceinline__ __bf16 f2bf(float f) {
  unsigned u = __builtin_bit_cast(unsigned, f);
  u += 0x7FFFu + ((u >> 16) & 1u);           // round-to-nearest-even
  unsigned short s = (unsigned short)(u >> 16);
  return __builtin_bit_cast(__bf16, s);
}

// ---- CDNA5 async global->LDS copy (16B per lane), tracked by ASYNCcnt ----
__device__ __forceinline__ void async_ld16(const void* lds, const void* g) {
  unsigned l = (unsigned)(unsigned long long)lds;   // low 32 bits = LDS offset
  asm volatile("global_load_async_to_lds_b128 %0, %1, off"
               :: "v"(l), "v"(g) : "memory");
}

// ---------------- generic f32 -> bf16 conversion ----------------
__global__ __launch_bounds__(256) void cvt_f32_bf16(
    const float* __restrict__ src, __bf16* __restrict__ dst, int n) {
  int i = blockIdx.x * 256 + threadIdx.x;
  if (i < n) dst[i] = f2bf(src[i]);
}

__global__ __launch_bounds__(256) void fill_zero_bf16(__bf16* __restrict__ p, int n) {
  int i = blockIdx.x * 256 + threadIdx.x;
  if (i < n) p[i] = __builtin_bit_cast(__bf16, (unsigned short)0);
}

// ---------------- LayerNorm (one block per token), emits bf16 ----------------
__global__ __launch_bounds__(256) void layernorm_bf16(
    const float* __restrict__ x, const float* __restrict__ g,
    const float* __restrict__ b, __bf16* __restrict__ xn) {
  __shared__ float red[256];
  int row = blockIdx.x, tid = threadIdx.x;
  const float* xr = x + (size_t)row * DM;
  float s = 0.f;
  for (int i = tid; i < DM; i += 256) s += xr[i];
  red[tid] = s; __syncthreads();
  for (int o = 128; o > 0; o >>= 1) { if (tid < o) red[tid] += red[tid + o]; __syncthreads(); }
  float mu = red[0] * (1.f / DM);
  __syncthreads();
  float v = 0.f;
  for (int i = tid; i < DM; i += 256) { float d = xr[i] - mu; v += d * d; }
  red[tid] = v; __syncthreads();
  for (int o = 128; o > 0; o >>= 1) { if (tid < o) red[tid] += red[tid + o]; __syncthreads(); }
  float rstd = rsqrtf(red[0] * (1.f / DM) + 1e-5f);
  __bf16* orow = xn + (size_t)row * DM;
  for (int i = tid; i < DM; i += 256)
    orow[i] = f2bf((xr[i] - mu) * rstd * g[i] + b[i]);
}

// ---------------- bf16 WMMA GEMM: C[M,N] = A[M,K] @ Bw[N,K]^T ----------------
// Block tile 128(M) x 256(N), 256 threads = 8 waves (wave32) as 2x4,
// wave tile 64x64 (16 WMMA accumulators).
// Double-buffered LDS staged with global_load_async_to_lds_b128 (ASYNCcnt):
// steady state issues next tile (6 async b128/thread) during current compute.
// Bw must be valid up to ceil(N/256)*256 rows (pad rows zeroed).
// epi: 0 = plain store, 1 = +bias then softplus, 2 = +residual
__global__ __launch_bounds__(256) void gemm_bf16_wmma(
    const __bf16* __restrict__ A, const __bf16* __restrict__ Bw,
    float* __restrict__ C, int M, int N, int K,
    int epi, const float* __restrict__ bias, const float* __restrict__ res) {
  __shared__ __bf16 sA[2][128 * 32];
  __shared__ __bf16 sB[2][256 * 32];
  const int tid   = threadIdx.x;
  const int mtile = blockIdx.y * 128, ntile = blockIdx.x * 256;
  const int w  = tid >> 5, lane = tid & 31;
  const int wm = w & 1,  wn = w >> 1;          // 2 x 4 wave grid
  const int lm = lane & 15, lhi = lane >> 4;

  Acc8 acc[4][4];
#pragma unroll
  for (int mf = 0; mf < 4; mf++)
#pragma unroll
    for (int nf = 0; nf < 4; nf++)
#pragma unroll
      for (int r = 0; r < 8; r++) acc[mf][nf].f[r] = 0.f;

  // A staging: 128 rows x 32 cols, 2 threads/row, 2 x 16B per thread
  const int lrow = tid >> 1;                   // 0..127
  const int lcol = (tid & 1) * 16;             // 0 or 16 (bf16 elems)
  const __bf16* Ag = A  + (size_t)(mtile + lrow) * K + lcol;
  const int aoff = lrow * 32 + lcol;
  // B staging: 256 rows x 32 cols, 1 thread/row, 4 x 16B per thread
  const __bf16* Bg = Bw + (size_t)(ntile + tid) * K;
  const int boff = tid * 32;

  auto issue = [&](int kt, int b) {
    const __bf16* ag = Ag + (kt << 5);
    const __bf16* bg = Bg + (kt << 5);
    async_ld16(&sA[b][aoff],      ag);
    async_ld16(&sA[b][aoff + 8],  ag + 8);
    async_ld16(&sB[b][boff],      bg);
    async_ld16(&sB[b][boff + 8],  bg + 8);
    async_ld16(&sB[b][boff + 16], bg + 16);
    async_ld16(&sB[b][boff + 24], bg + 24);
  };

  auto compute = [&](int b) {
    const __bf16* cA = sA[b];
    const __bf16* cB = sB[b];
    // B fragments: lane holds col n = lm, K = lhi*16 .. +15 (contiguous)
    Frag16 bfv[4];
#pragma unroll
    for (int nf = 0; nf < 4; nf++) {
      const __bf16* p = &cB[(wn * 64 + nf * 16 + lm) * 32 + lhi * 16];
      bfv[nf].h[0] = *(const v8bf*)p;
      bfv[nf].h[1] = *(const v8bf*)(p + 8);
    }
    // A fragments: lane holds row m = lm, K = lhi*16 .. +15 (contiguous)
#pragma unroll
    for (int mf = 0; mf < 4; mf++) {
      Frag16 af;
      const __bf16* p = &cA[(wm * 64 + mf * 16 + lm) * 32 + lhi * 16];
      af.h[0] = *(const v8bf*)p;
      af.h[1] = *(const v8bf*)(p + 8);
#pragma unroll
      for (int nf = 0; nf < 4; nf++)
        acc[mf][nf].v = __builtin_amdgcn_wmma_f32_16x16x32_bf16(
            false, af.v, false, bfv[nf].v, (short)0, acc[mf][nf].v, false, false);
    }
  };

  const int T = K >> 5;
  issue(0, 0);
  for (int i = 0; i < T - 1; i++) {            // steady state: overlap next loads
    issue(i + 1, (i + 1) & 1);
    asm volatile("s_wait_asynccnt 6" ::: "memory");  // tile i (first 6) landed
    __syncthreads();
    compute(i & 1);
    __syncthreads();
  }
  asm volatile("s_wait_asynccnt 0" ::: "memory");    // last tile landed
  __syncthreads();
  compute((T - 1) & 1);

  // store: C/D layout -> VGPR r: m = r + lhi*8, n = lm
#pragma unroll
  for (int mf = 0; mf < 4; mf++) {
#pragma unroll
    for (int nf = 0; nf < 4; nf++) {
      int col = ntile + wn * 64 + nf * 16 + lm;
      if (col < N) {
#pragma unroll
        for (int r = 0; r < 8; r++) {
          int row = mtile + wm * 64 + mf * 16 + lhi * 8 + r;
          float v = acc[mf][nf].f[r];
          if (epi == 1) { v += bias[col]; v = (v > 20.f) ? v : log1pf(__expf(v)); }
          else if (epi == 2) { v += res[(size_t)row * N + col]; }
          C[(size_t)row * N + col] = v;
        }
      }
    }
  }
}

// ---------------- causal depthwise conv1d (K=4) + bias + SiLU ----------------
__global__ __launch_bounds__(256) void conv_silu(
    const float* __restrict__ xz, const float* __restrict__ cw,
    const float* __restrict__ cb, float* __restrict__ xcf,
    __bf16* __restrict__ xcb) {
  int idx = blockIdx.x * 256 + threadIdx.x;
  if (idx >= BL * DI) return;
  int d = idx & (DI - 1);
  int row = idx >> 11;           // token index (b*L + l)
  int l = row & (LL - 1);
  const float* base = xz + (size_t)row * (2 * DI) + d;   // x_in = xz[:, :DI]
  float acc = cb[d];
#pragma unroll
  for (int j = 0; j < 4; j++) {
    int lo = l - 3 + j;
    if (lo >= 0) acc += cw[d * 4 + j] * base[(long)(j - 3) * (2 * DI)];
  }
  float s = acc / (1.f + __expf(-acc));
  xcf[idx] = s;
  xcb[idx] = f2bf(s);
}

// ---------------- slice dt_in (first 64 cols of x_dbl) to bf16 ----------------
__global__ __launch_bounds__(256) void slice_dtin(
    const float* __restrict__ xdbl, __bf16* __restrict__ dtin) {
  int idx = blockIdx.x * 256 + threadIdx.x;
  if (idx >= BL * DTR) return;
  int r = idx >> 6, c = idx & 63;
  dtin[idx] = f2bf(xdbl[(size_t)r * XPR + c]);
}

// ---------------- sequential selective scan, fused D-skip + SiLU(z) gate ----
// one lane per (batch, channel); 16-state recurrence in registers
__global__ __launch_bounds__(256) void ssm_scan(
    const float* __restrict__ delta, const float* __restrict__ xconv,
    const float* __restrict__ xdbl, const float* __restrict__ xz,
    const float* __restrict__ A_log, const float* __restrict__ Dp,
    __bf16* __restrict__ ygate) {
  int gid = blockIdx.x * 256 + threadIdx.x;
  if (gid >= BB * DI) return;
  int b = gid / DI, d = gid % DI;
  float Av[NST], h[NST];
#pragma unroll
  for (int n = 0; n < NST; n++) { Av[n] = -__expf(A_log[d * NST + n]); h[n] = 0.f; }
  float Dd = Dp[d];
  for (int l = 0; l < LL; l++) {
    int row = b * LL + l;
    float dt = delta[(size_t)row * DI + d];
    float xc = xconv[(size_t)row * DI + d];
    float zv = xz[(size_t)row * (2 * DI) + DI + d];    // z = xz[:, DI:]
    const float* bc = xdbl + (size_t)row * XPR;        // [dt(64) | B(16) | C(16)]
    float dx = dt * xc;
    float y = 0.f;
#pragma unroll
    for (int n = 0; n < NST; n++) {
      float dA = __expf(dt * Av[n]);
      h[n] = dA * h[n] + dx * bc[DTR + n];
      y += h[n] * bc[DTR + NST + n];
    }
    float gate = (y + xc * Dd) * (zv / (1.f + __expf(-zv)));
    ygate[(size_t)row * DI + d] = f2bf(gate);
  }
}

// ---------------- launcher ----------------
extern "C" void kernel_launch(void* const* d_in, const int* in_sizes, int n_in,
                              void* d_out, int out_size, void* d_ws, size_t ws_size,
                              hipStream_t stream) {
  (void)in_sizes; (void)n_in; (void)out_size; (void)ws_size;
  const float* x        = (const float*)d_in[0];
  const float* in_projw = (const float*)d_in[1];
  const float* conv_w   = (const float*)d_in[2];
  const float* conv_b   = (const float*)d_in[3];
  const float* x_projw  = (const float*)d_in[4];
  const float* dt_projw = (const float*)d_in[5];
  const float* dt_projb = (const float*)d_in[6];
  const float* A_log    = (const float*)d_in[7];
  const float* D_param  = (const float*)d_in[8];
  const float* out_projw= (const float*)d_in[9];
  const float* ln_g     = (const float*)d_in[10];
  const float* ln_b     = (const float*)d_in[11];
  float* out = (float*)d_out;

  // workspace carve-up (256B aligned)
  char* p = (char*)d_ws;
  auto carve = [&](size_t bytes) -> char* {
    char* r = p; p += (bytes + 255) & ~(size_t)255; return r;
  };
  __bf16* w_in   = (__bf16*)carve((size_t)2 * DI * DM * 2);   // [4096,1024] bf16
  __bf16* w_xp   = (__bf16*)carve((size_t)256 * DI * 2);      // [96->256,2048] padded
  __bf16* w_dt   = (__bf16*)carve((size_t)DI * DTR * 2);      // [2048,64]
  __bf16* w_out  = (__bf16*)carve((size_t)DM * DI * 2);       // [1024,2048]
  __bf16* xn     = (__bf16*)carve((size_t)BL * DM * 2);       // ln output
  float*  xz     = (float*) carve((size_t)BL * 2 * DI * 4);   // [4096,4096]
  float*  xcf    = (float*) carve((size_t)BL * DI * 4);       // conv f32
  __bf16* xcb    = (__bf16*)carve((size_t)BL * DI * 2);       // conv bf16
  float*  xdbl   = (float*) carve((size_t)BL * XPR * 4);      // [4096,96]
  __bf16* dtin   = (__bf16*)carve((size_t)BL * DTR * 2);      // [4096,64]
  float*  delta  = (float*) carve((size_t)BL * DI * 4);       // [4096,2048]
  __bf16* ygate  = (__bf16*)carve((size_t)BL * DI * 2);       // gated y bf16

  // 1) weight conversions (+ zero-pad rows 96..255 of x_proj copy)
  {
    int n1 = 2 * DI * DM, n2 = XPR * DI, n3 = DI * DTR, n4 = DM * DI;
    cvt_f32_bf16<<<(n1 + 255) / 256, 256, 0, stream>>>(in_projw, w_in, n1);
    cvt_f32_bf16<<<(n2 + 255) / 256, 256, 0, stream>>>(x_projw, w_xp, n2);
    fill_zero_bf16<<<((256 - XPR) * DI) / 256, 256, 0, stream>>>(
        w_xp + (size_t)XPR * DI, (256 - XPR) * DI);
    cvt_f32_bf16<<<(n3 + 255) / 256, 256, 0, stream>>>(dt_projw, w_dt, n3);
    cvt_f32_bf16<<<(n4 + 255) / 256, 256, 0, stream>>>(out_projw, w_out, n4);
  }
  // 2) LayerNorm
  layernorm_bf16<<<BL, 256, 0, stream>>>(x, ln_g, ln_b, xn);
  // 3) xz = xn @ in_proj^T   (4096 x 4096 x 1024)
  gemm_bf16_wmma<<<dim3((2 * DI) / 256, BL / 128), 256, 0, stream>>>(
      xn, w_in, xz, BL, 2 * DI, DM, 0, nullptr, nullptr);
  // 4) depthwise conv + SiLU
  conv_silu<<<(BL * DI) / 256, 256, 0, stream>>>(xz, conv_w, conv_b, xcf, xcb);
  // 5) x_dbl = x_conv @ x_proj^T   (4096 x 96 x 2048)
  gemm_bf16_wmma<<<dim3(1, BL / 128), 256, 0, stream>>>(
      xcb, w_xp, xdbl, BL, XPR, DI, 0, nullptr, nullptr);
  // 6) dt slice -> bf16, then delta = softplus(dt_in @ dt_proj^T + b)
  slice_dtin<<<(BL * DTR) / 256, 256, 0, stream>>>(xdbl, dtin);
  gemm_bf16_wmma<<<dim3(DI / 256, BL / 128), 256, 0, stream>>>(
      dtin, w_dt, delta, BL, DI, DTR, 1, dt_projb, nullptr);
  // 7) selective scan (fused gate)
  ssm_scan<<<(BB * DI) / 256, 256, 0, stream>>>(
      delta, xcf, xdbl, xz, A_log, D_param, ygate);
  // 8) out = ygate @ out_proj^T + residual   (4096 x 1024 x 2048)
  gemm_bf16_wmma<<<dim3(DM / 256, BL / 128), 256, 0, stream>>>(
      ygate, w_out, out, BL, DM, DI, 2, nullptr, x);
}
